// DSAModule_31559419691171
// MI455X (gfx1250) — matrix operand order repaired and checked by reading, not compiled, not run on hardware
//
#include <hip/hip_runtime.h>
#include <hip/hip_bf16.h>
#include <math.h>

// ---------------------------------------------------------------------------
// DSA forward for MI455X (gfx1250, wave32, WMMA).
// Heavy GEMMs (deform-conv main convs as implicit im2col GEMM) run on
// v_wmma_f32_16x16x32_f16 with fp32 accumulation. Offsets / masks / norms
// stay fp32 for accuracy (they feed floor() and normalization).
// ---------------------------------------------------------------------------

typedef __attribute__((ext_vector_type(16))) _Float16 v16h;
typedef __attribute__((ext_vector_type(8)))  float    v8f;

static constexpr int B_   = 2;
static constexpr int C_   = 64;
static constexpr int H_   = 128;
static constexpr int W_   = 128;
static constexpr int HW_  = H_ * W_;
static constexpr int M_   = B_ * HW_;     // 32768 GEMM rows (pixels)
static constexpr int NT_  = 9;            // deform taps

union H2 { _Float16 h[2]; unsigned u; };

__device__ __forceinline__ float sigmoidf_(float x) { return 1.0f / (1.0f + __expf(-x)); }

// ---------------------------------------------------------------------------
// 1) Convert main-conv weights (Cout, Cin, 3, 3) fp32 -> f16 [Cout][n*Cin + c]
//    K-ordering matches the im2col buffer produced by deform_sample_kernel.
// ---------------------------------------------------------------------------
__global__ void wconvert_kernel(const float* __restrict__ wmain,
                                _Float16* __restrict__ Wb,
                                int Cout, int Cin) {
  int idx = blockIdx.x * blockDim.x + threadIdx.x;
  int total = Cout * Cin * 9;
  if (idx >= total) return;
  int n = idx % 9;
  int t = idx / 9;
  int c = t % Cin;
  int o = t / Cin;
  Wb[(size_t)o * (9 * Cin) + (size_t)n * Cin + c] = (_Float16)wmain[idx];
}

// ---------------------------------------------------------------------------
// 2) Offset (2N ch) + mask (N ch, sigmoid fused) 3x3 convs, direct fp32.
//    One thread per output pixel; 27 accumulators; weights hit L0/L2.
// ---------------------------------------------------------------------------
__global__ __launch_bounds__(256)
void offmask_conv_kernel(const float* __restrict__ x,    // [B,Cin,H,W]
                         const float* __restrict__ wp,   // [18,Cin,3,3]
                         const float* __restrict__ bp,   // [18]
                         const float* __restrict__ wm,   // [9,Cin,3,3]
                         const float* __restrict__ bm,   // [9]
                         float* __restrict__ off,        // [B,18,H,W]
                         float* __restrict__ mask,       // [B,9,H,W] (sigmoid)
                         int Cin) {
  int idx = blockIdx.x * blockDim.x + threadIdx.x;
  if (idx >= B_ * HW_) return;
  int b  = idx / HW_;
  int hw = idx - b * HW_;
  int h  = hw / W_;
  int w  = hw - h * W_;

  float accO[18];
  float accM[9];
#pragma unroll
  for (int o = 0; o < 18; ++o) accO[o] = bp[o];
#pragma unroll
  for (int o = 0; o < 9; ++o)  accM[o] = bm[o];

  for (int c = 0; c < Cin; ++c) {
    const float* xc = x + ((size_t)b * Cin + c) * HW_;
    float t[9];
#pragma unroll
    for (int n = 0; n < 9; ++n) {
      int yy = h + n / 3 - 1;
      int xx = w + n % 3 - 1;
      t[n] = (yy >= 0 && yy < H_ && xx >= 0 && xx < W_) ? xc[yy * W_ + xx] : 0.0f;
    }
#pragma unroll
    for (int o = 0; o < 18; ++o) {
      const float* wr = wp + ((size_t)o * Cin + c) * 9;
      float s = 0.0f;
#pragma unroll
      for (int n = 0; n < 9; ++n) s = fmaf(wr[n], t[n], s);
      accO[o] += s;
    }
#pragma unroll
    for (int o = 0; o < 9; ++o) {
      const float* wr = wm + ((size_t)o * Cin + c) * 9;
      float s = 0.0f;
#pragma unroll
      for (int n = 0; n < 9; ++n) s = fmaf(wr[n], t[n], s);
      accM[o] += s;
    }
  }
#pragma unroll
  for (int o = 0; o < 18; ++o) off[((size_t)b * 18 + o) * HW_ + hw] = accO[o];
#pragma unroll
  for (int o = 0; o < 9;  ++o) mask[((size_t)b * 9 + o) * HW_ + hw] = sigmoidf_(accM[o]);
}

// ---------------------------------------------------------------------------
// 3) Bilinear sampling + mask modulation -> f16 im2col A[M][9*Cin],
//    k = n*Cin + c. One thread per (pixel, tap); loops channels in pairs
//    for 4-byte packed stores.
// ---------------------------------------------------------------------------
__global__ __launch_bounds__(256)
void deform_sample_kernel(const float* __restrict__ x,     // [B,Cin,H,W]
                          const float* __restrict__ off,   // [B,18,H,W]
                          const float* __restrict__ mask,  // [B,9,H,W]
                          _Float16* __restrict__ A,        // [M][9*Cin]
                          int Cin) {
  int idx = blockIdx.x * blockDim.x + threadIdx.x;
  if (idx >= M_ * NT_) return;
  int n  = idx % NT_;
  int p  = idx / NT_;
  int b  = p / HW_;
  int hw = p - b * HW_;
  int h  = hw / W_;
  int w  = hw - h * W_;
  const int Hp = H_ + 2, Wp = W_ + 2;

  float ox = off[((size_t)b * 18 + n) * HW_ + hw];
  float oy = off[((size_t)b * 18 + 9 + n) * HW_ + hw];
  float mv = mask[((size_t)b * 9 + n) * HW_ + hw];

  float px = ox + (float)(n / 3 - 1) + (float)(h + 1);
  float py = oy + (float)(n % 3 - 1) + (float)(w + 1);
  float fx = floorf(px), fy = floorf(py);

  int qlx = min(max((int)fx, 0), Hp - 1);
  int qly = min(max((int)fy, 0), Wp - 1);
  int qrx = min(max((int)fx + 1, 0), Hp - 1);
  int qry = min(max((int)fy + 1, 0), Wp - 1);
  px = fminf(fmaxf(px, 0.0f), (float)(Hp - 1));
  py = fminf(fmaxf(py, 0.0f), (float)(Wp - 1));

  float glt = (1.0f + ((float)qlx - px)) * (1.0f + ((float)qly - py));
  float grb = (1.0f - ((float)qrx - px)) * (1.0f - ((float)qry - py));
  float glb = (1.0f + ((float)qlx - px)) * (1.0f - ((float)qry - py));
  float grt = (1.0f - ((float)qrx - px)) * (1.0f + ((float)qly - py));

  // padded-image -> interior mapping (pad row/col of zeros around x)
  bool v00 = (qlx >= 1 && qlx <= H_ && qly >= 1 && qly <= W_); // (lt)
  bool v11 = (qrx >= 1 && qrx <= H_ && qry >= 1 && qry <= W_); // (rb)
  bool v01 = (qlx >= 1 && qlx <= H_ && qry >= 1 && qry <= W_); // (lb)
  bool v10 = (qrx >= 1 && qrx <= H_ && qly >= 1 && qly <= W_); // (rt)
  int i00 = (qlx - 1) * W_ + (qly - 1);
  int i11 = (qrx - 1) * W_ + (qry - 1);
  int i01 = (qlx - 1) * W_ + (qry - 1);
  int i10 = (qrx - 1) * W_ + (qly - 1);

  const float* xb = x + (size_t)b * Cin * HW_;
  _Float16* arow = A + (size_t)p * (9 * Cin) + (size_t)n * Cin;

  for (int c = 0; c < Cin; c += 2) {
    const float* xc0 = xb + (size_t)c * HW_;
    const float* xc1 = xc0 + HW_;
    float s0 = 0.0f, s1 = 0.0f;
    if (v00) { s0 = fmaf(glt, xc0[i00], s0); s1 = fmaf(glt, xc1[i00], s1); }
    if (v11) { s0 = fmaf(grb, xc0[i11], s0); s1 = fmaf(grb, xc1[i11], s1); }
    if (v01) { s0 = fmaf(glb, xc0[i01], s0); s1 = fmaf(glb, xc1[i01], s1); }
    if (v10) { s0 = fmaf(grt, xc0[i10], s0); s1 = fmaf(grt, xc1[i10], s1); }
    H2 t;
    t.h[0] = (_Float16)(s0 * mv);
    t.h[1] = (_Float16)(s1 * mv);
    *reinterpret_cast<unsigned*>(arow + c) = t.u;
  }
}

// ---------------------------------------------------------------------------
// 4) WMMA GEMM: Cout[n][m] = sum_k A[m][k] * Wb[n][k]
//    Block = 128 threads (4 waves); block -> one 16-row M tile; each wave
//    owns one 16-col N tile. f16 operands, f32 accumulate.
//    Fragment layouts per CDNA5 ISA 7.12.2 (16-bit A 16x32, B 32x16, f32 D).
// ---------------------------------------------------------------------------
__global__ __launch_bounds__(128)
void gemm_wmma_kernel(const _Float16* __restrict__ A,   // [M][K]
                      const _Float16* __restrict__ Wb,  // [N][K]
                      float* __restrict__ Cout,         // [N][M]
                      int M, int N, int K) {
  const int wave = threadIdx.x >> 5;
  const int lane = threadIdx.x & 31;
  const int mt   = blockIdx.x;
  const int n0   = wave * 16;
  if (n0 >= N) return;
  const int m0  = mt * 16;
  const int l15 = lane & 15;
  const int hi  = lane >> 4;

  // A: lane row m0+l15; element e<8 -> K=kc+hi*8+e ; e>=8 -> K=kc+16+hi*8+(e-8)
  const _Float16* aRow = A  + (size_t)(m0 + l15) * K + hi * 8;
  // B: lane col n0+l15; element e<8 -> K=kc+hi*8+e ; e>=8 -> K=kc+16+hi*8+(e-8)
  const _Float16* bRow = Wb + (size_t)(n0 + l15) * K + hi * 8;

  v8f acc = {};
  for (int kc = 0; kc < K; kc += 32) {
    v16h a, b;
    reinterpret_cast<uint4*>(&a)[0] = *reinterpret_cast<const uint4*>(aRow + kc);
    reinterpret_cast<uint4*>(&a)[1] = *reinterpret_cast<const uint4*>(aRow + kc + 16);
    reinterpret_cast<uint4*>(&b)[0] = *reinterpret_cast<const uint4*>(bRow + kc);
    reinterpret_cast<uint4*>(&b)[1] = *reinterpret_cast<const uint4*>(bRow + kc + 16);
    acc = __builtin_amdgcn_wmma_f32_16x16x32_f16(false, a, false, b,
                                                 (short)0, acc, false, false);
  }
  // D: lane col n0+l15; element v -> row m0 + hi*8 + v  (8 contiguous rows)
  float* out = Cout + (size_t)(n0 + l15) * M + m0 + hi * 8;
  float4 o0 = { acc[0], acc[1], acc[2], acc[3] };
  float4 o1 = { acc[4], acc[5], acc[6], acc[7] };
  *reinterpret_cast<float4*>(out)     = o0;
  *reinterpret_cast<float4*>(out + 4) = o1;
}

// ---------------------------------------------------------------------------
// 5) GroupNorm + ReLU. src is GEMM layout [Cout][b*HW+hw]; dst is (B,C,H,W).
//    One block per (group, batch); 256-thread LDS tree reduction.
// ---------------------------------------------------------------------------
__global__ __launch_bounds__(256)
void groupnorm_relu_kernel(const float* __restrict__ src,   // [Cout][M]
                           const float* __restrict__ gamma,
                           const float* __restrict__ beta,
                           float* __restrict__ dst,          // [B,Cout,HW]
                           int Cout, int chPerGroup, float eps) {
  const int g   = blockIdx.x;
  const int b   = blockIdx.y;
  const int c0  = g * chPerGroup;
  const int tid = threadIdx.x;

  float sum = 0.0f, sq = 0.0f;
  for (int c = 0; c < chPerGroup; ++c) {
    const float* s = src + (size_t)(c0 + c) * M_ + (size_t)b * HW_;
    for (int i = tid; i < HW_; i += blockDim.x) {
      float v = s[i];
      sum += v;
      sq  = fmaf(v, v, sq);
    }
  }
  __shared__ float rs[256];
  __shared__ float rq[256];
  rs[tid] = sum; rq[tid] = sq;
  __syncthreads();
  for (int s = 128; s > 0; s >>= 1) {
    if (tid < s) { rs[tid] += rs[tid + s]; rq[tid] += rq[tid + s]; }
    __syncthreads();
  }
  const float cnt  = (float)(chPerGroup * HW_);
  const float mean = rs[0] / cnt;
  const float var  = rq[0] / cnt - mean * mean;
  const float inv  = rsqrtf(var + eps);

  for (int c = 0; c < chPerGroup; ++c) {
    const float ga = gamma[c0 + c] * inv;
    const float be = beta[c0 + c];
    const float* s = src + (size_t)(c0 + c) * M_ + (size_t)b * HW_;
    float* d = dst + ((size_t)b * Cout + c0 + c) * HW_;
    for (int i = tid; i < HW_; i += blockDim.x) {
      float v = fmaf(s[i] - mean, ga, be);
      d[i] = fmaxf(v, 0.0f);
    }
  }
}

// ---------------------------------------------------------------------------
// 6) Channel mean / max pooling over C channels -> [B,H,W] maps.
// ---------------------------------------------------------------------------
__global__ __launch_bounds__(256)
void chanpool_kernel(const float* __restrict__ c1,   // [B,C,HW]
                     float* __restrict__ avgm,       // [B,HW]
                     float* __restrict__ maxm) {     // [B,HW]
  int idx = blockIdx.x * blockDim.x + threadIdx.x;
  if (idx >= B_ * HW_) return;
  int b  = idx / HW_;
  int hw = idx - b * HW_;
  float s = 0.0f, mx = -INFINITY;
  for (int c = 0; c < C_; ++c) {
    float v = c1[((size_t)b * C_ + c) * HW_ + hw];
    s += v;
    mx = fmaxf(mx, v);
  }
  avgm[idx] = s * (1.0f / (float)C_);
  maxm[idx] = mx;
}

// ---------------------------------------------------------------------------
// 7) 3x3 attention convs on pooled maps + sigmoid, then concat-modulate:
//    cat[:, 0:C]  = sigmoid(conv(avg)) * c1
//    cat[:, C:2C] = sigmoid(conv(max)) * c1
// ---------------------------------------------------------------------------
__global__ __launch_bounds__(256)
void attn_apply_kernel(const float* __restrict__ c1,     // [B,C,HW]
                       const float* __restrict__ avgm,   // [B,HW]
                       const float* __restrict__ maxm,   // [B,HW]
                       const float* __restrict__ wa, const float* __restrict__ ba,
                       const float* __restrict__ wb, const float* __restrict__ bb,
                       float* __restrict__ cat) {        // [B,2C,HW]
  int idx = blockIdx.x * blockDim.x + threadIdx.x;
  if (idx >= B_ * HW_) return;
  int b  = idx / HW_;
  int hw = idx - b * HW_;
  int h  = hw / W_;
  int w  = hw - h * W_;

  float sa = ba[0], sb = bb[0];
#pragma unroll
  for (int n = 0; n < 9; ++n) {
    int yy = h + n / 3 - 1;
    int xx = w + n % 3 - 1;
    if (yy >= 0 && yy < H_ && xx >= 0 && xx < W_) {
      float av = avgm[(size_t)b * HW_ + yy * W_ + xx];
      float mv = maxm[(size_t)b * HW_ + yy * W_ + xx];
      sa = fmaf(wa[n], av, sa);
      sb = fmaf(wb[n], mv, sb);
    }
  }
  float ga = sigmoidf_(sa);
  float gm = sigmoidf_(sb);
  for (int c = 0; c < C_; ++c) {
    float v = c1[((size_t)b * C_ + c) * HW_ + hw];
    cat[((size_t)b * 2 * C_ + c) * HW_ + hw]      = ga * v;
    cat[((size_t)b * 2 * C_ + C_ + c) * HW_ + hw] = gm * v;
  }
}

// ---------------------------------------------------------------------------
// Host launcher
// ---------------------------------------------------------------------------
extern "C" void kernel_launch(void* const* d_in, const int* in_sizes, int n_in,
                              void* d_out, int out_size, void* d_ws, size_t ws_size,
                              hipStream_t stream) {
  (void)in_sizes; (void)n_in; (void)out_size; (void)ws_size;

  const float* x   = (const float*)d_in[0];
  const float* wp1 = (const float*)d_in[1];
  const float* bp1 = (const float*)d_in[2];
  const float* wm1 = (const float*)d_in[3];
  const float* bm1 = (const float*)d_in[4];
  const float* wc1 = (const float*)d_in[5];
  const float* g1  = (const float*)d_in[6];
  const float* be1 = (const float*)d_in[7];
  const float* wa  = (const float*)d_in[8];
  const float* ba  = (const float*)d_in[9];
  const float* wb  = (const float*)d_in[10];
  const float* bb  = (const float*)d_in[11];
  const float* wp2 = (const float*)d_in[12];
  const float* bp2 = (const float*)d_in[13];
  const float* wm2 = (const float*)d_in[14];
  const float* bm2 = (const float*)d_in[15];
  const float* wc2 = (const float*)d_in[16];
  const float* g2  = (const float*)d_in[17];
  const float* be2 = (const float*)d_in[18];
  float* out = (float*)d_out;

  // ---- workspace carve (all offsets 256B aligned) ----
  const size_t K1 = 9 * C_;        // 576
  const size_t K2 = 9 * 2 * C_;    // 1152
  size_t off_bytes = 0;
  auto carve = [&](size_t bytes) {
    size_t o = off_bytes;
    off_bytes += (bytes + 255) & ~(size_t)255;
    return (char*)d_ws + o;
  };
  _Float16* A       = (_Float16*)carve((size_t)M_ * K2 * sizeof(_Float16)); // shared by both stages
  float*    convbuf = (float*)   carve((size_t)C_ * M_ * sizeof(float));    // GEMM out [N][M]
  float*    c1      = (float*)   carve((size_t)B_ * C_ * HW_ * sizeof(float));
  float*    cat     = (float*)   carve((size_t)B_ * 2 * C_ * HW_ * sizeof(float));
  float*    offb    = (float*)   carve((size_t)B_ * 18 * HW_ * sizeof(float));
  float*    maskb   = (float*)   carve((size_t)B_ * 9 * HW_ * sizeof(float));
  float*    avgm    = (float*)   carve((size_t)B_ * HW_ * sizeof(float));
  float*    maxm    = (float*)   carve((size_t)B_ * HW_ * sizeof(float));
  _Float16* Wb1     = (_Float16*)carve((size_t)C_ * K1 * sizeof(_Float16));
  _Float16* Wb2     = (_Float16*)carve((size_t)C_ * K2 * sizeof(_Float16));

  const int TPB = 256;
  const int pixBlocks  = (B_ * HW_ + TPB - 1) / TPB;
  const int sampBlocks = (M_ * NT_ + TPB - 1) / TPB;

  // ---- weight conversions ----
  wconvert_kernel<<<(C_ * C_ * 9 + TPB - 1) / TPB, TPB, 0, stream>>>(wc1, Wb1, C_, C_);
  wconvert_kernel<<<(C_ * 2 * C_ * 9 + TPB - 1) / TPB, TPB, 0, stream>>>(wc2, Wb2, C_, 2 * C_);

  // ---- stage 1: deform conv on x (Cin = 64) ----
  offmask_conv_kernel<<<pixBlocks, TPB, 0, stream>>>(x, wp1, bp1, wm1, bm1, offb, maskb, C_);
  deform_sample_kernel<<<sampBlocks, TPB, 0, stream>>>(x, offb, maskb, A, C_);
  gemm_wmma_kernel<<<M_ / 16, 128, 0, stream>>>(A, Wb1, convbuf, M_, C_, (int)K1);
  groupnorm_relu_kernel<<<dim3(C_ / 4, B_), TPB, 0, stream>>>(convbuf, g1, be1, c1, C_, 4, 1e-5f);

  // ---- spatial attention -> cat (B, 2C, H, W) ----
  chanpool_kernel<<<pixBlocks, TPB, 0, stream>>>(c1, avgm, maxm);
  attn_apply_kernel<<<pixBlocks, TPB, 0, stream>>>(c1, avgm, maxm, wa, ba, wb, bb, cat);

  // ---- stage 2: deform conv on cat (Cin = 128) ----
  offmask_conv_kernel<<<pixBlocks, TPB, 0, stream>>>(cat, wp2, bp2, wm2, bm2, offb, maskb, 2 * C_);
  deform_sample_kernel<<<sampBlocks, TPB, 0, stream>>>(cat, offb, maskb, A, 2 * C_);
  gemm_wmma_kernel<<<M_ / 16, 128, 0, stream>>>(A, Wb2, convbuf, M_, C_, (int)K2);
  groupnorm_relu_kernel<<<dim3(C_ / 4, B_), TPB, 0, stream>>>(convbuf, g2, be2, out, C_, 4, 1e-5f);
}
